// MultiHeadedAttention_69114613729784
// MI455X (gfx1250) — compile-verified, matrix operand order checked by reading, not run on hardware
//
#include <hip/hip_runtime.h>

// ---------------------------------------------------------------------------
// MHA forward for MI455X (gfx1250): bf16 WMMA + TDM (tensor_load_to_lds) for
// all pure-copy LDS staging, flash-style attention.
// B=4, S=1024, D=1024, H=16, dk=64.
// ---------------------------------------------------------------------------

#define B_ 4
#define S_ 1024
#define D_ 1024
#define H_ 16
#define DK_ 64

typedef __attribute__((ext_vector_type(16))) __bf16 v16bf;
typedef __attribute__((ext_vector_type(8)))  float  v8f;
typedef __attribute__((ext_vector_type(4)))  unsigned int v4u;
typedef __attribute__((ext_vector_type(8)))  unsigned int v8u;

union FragBF {
    unsigned int u[8];
    uint4 q[2];
    v16bf v;
};

static __device__ inline v8f vzero() {
    v8f z = {0.f, 0.f, 0.f, 0.f, 0.f, 0.f, 0.f, 0.f};
    return z;
}

// hardware f32 -> bf16 (RNE)
static __device__ inline unsigned short f32_to_bf16(float f) {
    union { __bf16 h; unsigned short u; } c;
    c.h = (__bf16)f;
    return c.u;
}
static __device__ inline unsigned int pack_bf16x2(float lo, float hi) {
    union { __bf16 h[2]; unsigned int u; } c;
    c.h[0] = (__bf16)lo;
    c.h[1] = (__bf16)hi;
    return c.u;
}

// A-fragment (16x32 bf16, M x K): per lane two contiguous 8-element runs.
static __device__ inline void load_frag_a(FragBF& fa, const unsigned short* src, int ld) {
    const int lane = threadIdx.x & 31;
    const int m = lane & 15;
    const int hh = lane >> 4;
    const unsigned short* row = src + (size_t)m * ld + (hh << 3);
    fa.q[0] = *(const uint4*)(row);
    fa.q[1] = *(const uint4*)(row + 16);
}

// B-fragment from swizzled storage: one contiguous 16-ushort run per lane.
static __device__ inline void load_frag_b_sw(FragBF& fb, const unsigned short* swbase) {
    const int lane = threadIdx.x & 31;
    const unsigned short* p = swbase + (lane << 4);
    fb.q[0] = *(const uint4*)(p);
    fb.q[1] = *(const uint4*)(p + 8);
}

// swizzled B index: (k,n) -> ((n/16)*32 + n%16 + ((k>>3)&1)*16)*16 + (k>>4)*8 + k%8
static __device__ inline int bsw_seg_base(int n, int kseg) {
    return ((((n >> 4) << 5) + (n & 15) + ((kseg & 1) << 4)) << 4) + ((kseg >> 1) << 3);
}

static __device__ inline v8f wmma_bf16(const FragBF& a, const FragBF& b, v8f c) {
    return __builtin_amdgcn_wmma_f32_16x16x32_bf16(false, a.v, false, b.v,
                                                   (short)0, c, false, false);
}

// generic LDS pointer -> 32-bit LDS byte offset (low 32 bits of flat address)
static __device__ inline unsigned int lds_offset(const void* p) {
    return (unsigned int)(unsigned long long)(size_t)p;
}

// ---------------------------------------------------------------------------
// TDM: issue a 2D tile load (global bf16 -> LDS) via tensor_load_to_lds.
// D# per CDNA5 ISA: group0 = {count|flags, lds_addr, gaddr_lo, gaddr_hi|type},
// group1 = {dsize/pad, tensor dims, tile dims, dim0 stride}.  Groups 2/3 zero
// (2D tensor).  pad_enable inserts (pad_amount+1) dwords after every
// 2^(pad_interval+1) dwords -> padded LDS row stride.
// ---------------------------------------------------------------------------
static __device__ inline void tdm_load_2d(const void* gptr, unsigned int lds_off,
                                          unsigned int tensor_d0, unsigned int tensor_d1,
                                          unsigned long long stride0,
                                          unsigned int tile_d0, unsigned int tile_d1,
                                          unsigned int pad_interval, unsigned int pad_amount)
{
    unsigned long long ga = (unsigned long long)(size_t)gptr;
    v4u g0;
    g0[0] = 1u;                                                  // count=1 (valid, user)
    g0[1] = lds_off;                                             // lds_addr (bytes)
    g0[2] = (unsigned int)(ga & 0xffffffffu);                    // global_addr[31:0]
    g0[3] = (unsigned int)((ga >> 32) & 0x01ffffffu) | 0x80000000u;  // addr[56:32]|type=2
    v8u g1;
    g1[0] = (1u << 16)                                           // data_size = 2 bytes
          | (1u << 20)                                           // pad_enable
          | (pad_interval << 22) | (pad_amount << 25);
    g1[1] = (tensor_d0 & 0xffffu) << 16;                         // tensor_dim0[15:0]
    g1[2] = ((tensor_d0 >> 16) & 0xffffu) | ((tensor_d1 & 0xffffu) << 16);
    g1[3] = ((tensor_d1 >> 16) & 0xffffu) | ((tile_d0 & 0xffffu) << 16);
    g1[4] = (tile_d1 & 0xffffu);                                 // tile_dim1 (tile_dim2=0)
    g1[5] = (unsigned int)(stride0 & 0xffffffffu);               // dim0_stride[31:0]
    g1[6] = (unsigned int)((stride0 >> 32) & 0xffffu);           // dim0_stride[47:32]
    g1[7] = 0u;
    v4u gz = {0u, 0u, 0u, 0u};
    asm volatile("tensor_load_to_lds %0, %1, %2, %3"
                 :: "s"(g0), "s"(g1), "s"(gz), "s"(gz)
                 : "memory");
}

template <int N>
static __device__ inline void wait_tensorcnt() {
#if defined(__has_builtin)
#if __has_builtin(__builtin_amdgcn_s_wait_tensorcnt)
    __builtin_amdgcn_s_wait_tensorcnt((short)N);
#else
    asm volatile("s_wait_tensorcnt %0" :: "n"(N) : "memory");
#endif
#else
    asm volatile("s_wait_tensorcnt %0" :: "n"(N) : "memory");
#endif
}

// ---------------------------------------------------------------------------
// Kernel 1: QKV projections (f32 -> bf16 conversion path; TDM not applicable).
// Block tile 64(M) x 128(N), wave tile 32x32 (4 WMMA).
// q,k written head-major [B,H,S,DK]; v written TRANSPOSED per head [B,H,DK,S]
// so attention's V B-fragments become contiguous A-pattern reads.
// grid = (D/128, B*S/64, 3), block = 256.
// ---------------------------------------------------------------------------
__global__ __launch_bounds__(256) void qkv_proj_kernel(
    const float* __restrict__ Q, const float* __restrict__ K, const float* __restrict__ V,
    const float* __restrict__ Wq, const float* __restrict__ Wk, const float* __restrict__ Wv,
    unsigned short* __restrict__ qo, unsigned short* __restrict__ ko,
    unsigned short* __restrict__ vo)
{
    __shared__ __align__(16) unsigned short As[64][40];
    __shared__ __align__(16) unsigned short Bsw[4096];

    const float* X = (blockIdx.z == 0) ? Q : (blockIdx.z == 1) ? K : V;
    const float* W = (blockIdx.z == 0) ? Wq : (blockIdx.z == 1) ? Wk : Wv;
    unsigned short* O = (blockIdx.z == 0) ? qo : (blockIdx.z == 1) ? ko : vo;
    const bool transpose_out = (blockIdx.z == 2);

    const int m0 = blockIdx.y * 64;
    const int n0 = blockIdx.x * 128;
    const int tid = threadIdx.x;
    const int lane = tid & 31;
    const int wave = tid >> 5;
    const int wm = wave >> 2;
    const int wn = wave & 3;

    const int arow = tid >> 2;
    const int ac0 = (tid & 3) * 8;

    v8f acc[2][2];
#pragma unroll
    for (int i = 0; i < 2; ++i)
#pragma unroll
        for (int j = 0; j < 2; ++j) acc[i][j] = vzero();

    for (int k0 = 0; k0 < D_; k0 += 32) {
        {   // A tile: 64x32 f32 -> bf16
            const float* src = X + (size_t)(m0 + arow) * D_ + k0 + ac0;
            float4 f0 = *(const float4*)(src);
            float4 f1 = *(const float4*)(src + 4);
            uint4 pk;
            pk.x = pack_bf16x2(f0.x, f0.y);
            pk.y = pack_bf16x2(f0.z, f0.w);
            pk.z = pack_bf16x2(f1.x, f1.y);
            pk.w = pack_bf16x2(f1.z, f1.w);
            *(uint4*)&As[arow][ac0] = pk;
            if (k0 + 32 < D_) __builtin_prefetch(src + 32, 0, 3);
        }
        {   // B tile: 32x128 f32 -> bf16 swizzled
#pragma unroll
            for (int s2 = 0; s2 < 2; ++s2) {
                int seg = tid + s2 * 256;
                int n = seg & 127;
                int kseg = seg >> 7;
                const float* src = W + (size_t)(k0 + kseg * 8) * D_ + n0 + n;
                uint4 pk;
                pk.x = pack_bf16x2(src[0 * D_], src[1 * D_]);
                pk.y = pack_bf16x2(src[2 * D_], src[3 * D_]);
                pk.z = pack_bf16x2(src[4 * D_], src[5 * D_]);
                pk.w = pack_bf16x2(src[6 * D_], src[7 * D_]);
                *(uint4*)(Bsw + bsw_seg_base(n, kseg)) = pk;
                if (k0 + 32 < D_) __builtin_prefetch(src + 32 * D_, 0, 3);
            }
        }
        __syncthreads();

        FragBF a0, a1, b0, b1;
        load_frag_a(a0, &As[wm * 32][0], 40);
        load_frag_a(a1, &As[wm * 32 + 16][0], 40);
        load_frag_b_sw(b0, Bsw + (wn * 2 + 0) * 512);
        load_frag_b_sw(b1, Bsw + (wn * 2 + 1) * 512);
        acc[0][0] = wmma_bf16(a0, b0, acc[0][0]);
        acc[0][1] = wmma_bf16(a0, b1, acc[0][1]);
        acc[1][0] = wmma_bf16(a1, b0, acc[1][0]);
        acc[1][1] = wmma_bf16(a1, b1, acc[1][1]);
        __syncthreads();
    }

    const int mbase = ((lane >> 4) << 3);
#pragma unroll
    for (int ni = 0; ni < 2; ++ni) {
        int nn = n0 + wn * 32 + ni * 16 + (lane & 15);
        int head = nn >> 6;
        int dd = nn & 63;
#pragma unroll
        for (int mi = 0; mi < 2; ++mi)
#pragma unroll
            for (int i = 0; i < 8; ++i) {
                int r = m0 + wm * 32 + mi * 16 + mbase + i;
                int bb = r >> 10, ss = r & 1023;
                size_t idx;
                if (transpose_out)   // V^T: [B,H,DK,S]
                    idx = (((size_t)(bb * H_ + head) * DK_ + dd) << 10) + ss;
                else                 // Q,K: [B,H,S,DK]
                    idx = (((size_t)(bb * H_ + head) << 10) + ss) * DK_ + dd;
                O[idx] = f32_to_bf16(acc[mi][ni][i]);
            }
    }
}

// ---------------------------------------------------------------------------
// Kernel 2: flash attention per (b,h).  Wave = 16 query rows, block = 128.
// K and V^T tiles double-buffered in LDS, staged by TDM (wave 0 issues,
// TENSORcnt-pipelined: tile t+1 DMA overlaps tile t compute).
// grid = (S/128, B*H), block = 256.
// ---------------------------------------------------------------------------
__global__ __launch_bounds__(256) void attn_kernel(
    const unsigned short* __restrict__ qw, const unsigned short* __restrict__ kw,
    const unsigned short* __restrict__ vw, const int* __restrict__ mask,
    unsigned short* __restrict__ ctx)
{
    __shared__ __align__(16) unsigned short Ks[2][32][72];   // K: 32 x 64 (+pad)
    __shared__ __align__(16) unsigned short VTs[2][64][40];  // V^T: 64 x 32 (+pad)
    __shared__ __align__(16) unsigned short Ps[8][16][40];

    const float NEG_INF = -__builtin_inff();

    const int bh = blockIdx.y;
    const int b = bh >> 4;
    const int h = bh & 15;
    const int tid = threadIdx.x;
    const int wave = tid >> 5;
    const int lane = tid & 31;
    const int n = lane & 15;
    const int hh = lane >> 4;
    const int q0 = blockIdx.x * 128 + wave * 16;

    const unsigned short* qbase = qw + (((size_t)bh) << 10) * DK_;    // [S][DK]
    const unsigned short* kbase = kw + (((size_t)bh) << 10) * DK_;    // [S][DK]
    const unsigned short* vtbase = vw + (((size_t)bh * DK_) << 10);   // [DK][S]

    FragBF qa0, qa1;
    load_frag_a(qa0, qbase + (size_t)q0 * DK_ + 0, DK_);
    load_frag_a(qa1, qbase + (size_t)q0 * DK_ + 32, DK_);

    float mrun[8], lrun[8];
    v8f oacc[4];
#pragma unroll
    for (int i = 0; i < 8; ++i) { mrun[i] = NEG_INF; lrun[i] = 0.f; }
#pragma unroll
    for (int t = 0; t < 4; ++t) oacc[t] = vzero();

    const int ntiles = (blockIdx.x * 128 + 128) / 32;  // block-uniform causal bound

    // K tile: rows=32 kv, row len 64 el = 128B = 32 dw; pad 4 dw -> stride 72 ush
    // VT tile: rows=64 d, row len 32 el = 64B = 16 dw; pad 4 dw -> stride 40 ush
    if (wave == 0) {
        tdm_load_2d(kbase, lds_offset(&Ks[0][0][0]), DK_, S_, DK_, DK_, 32, 4u, 3u);
        tdm_load_2d(vtbase, lds_offset(&VTs[0][0][0]), S_, DK_, S_, 32, DK_, 3u, 3u);
    }

    for (int t = 0; t < ntiles; ++t) {
        const int kv0 = t * 32;
        const int buf = t & 1;
        const bool has_next = (t + 1 < ntiles);
        if (wave == 0) {
            if (has_next) {   // issue tile t+1 into the other buffer (DMA overlap)
                const int kv1 = kv0 + 32;
                tdm_load_2d(kbase + (size_t)kv1 * DK_, lds_offset(&Ks[buf ^ 1][0][0]),
                            DK_, S_, DK_, DK_, 32, 4u, 3u);
                tdm_load_2d(vtbase + kv1, lds_offset(&VTs[buf ^ 1][0][0]),
                            S_, DK_, S_, 32, DK_, 3u, 3u);
                wait_tensorcnt<2>();   // tile t's 2 DMAs done (in-order)
            } else {
                wait_tensorcnt<0>();
            }
        }
        __syncthreads();   // releases all waves once tile t data is in LDS

        if (kv0 <= q0 + 15) {  // causal: wave has live rows in this tile
            // ---- scores: S = Q @ K^T ----
            float pv[2][8];
#pragma unroll
            for (int tt = 0; tt < 2; ++tt) {
                v8f sf = vzero();
                FragBF kb;
                load_frag_a(kb, &Ks[buf][tt * 16][0], 72);
                sf = wmma_bf16(qa0, kb, sf);
                load_frag_a(kb, &Ks[buf][tt * 16][32], 72);
                sf = wmma_bf16(qa1, kb, sf);

                const int j = kv0 + tt * 16 + n;
#pragma unroll
                for (int i = 0; i < 8; ++i) {
                    int qr = q0 + hh * 8 + i;
                    float raw = sf[i] * 0.125f;  // 1/sqrt(64)
                    int mv = mask[((size_t)(b * S_ + qr) << 10) + j];
                    float val = (mv != 0 && j <= qr) ? raw : 0.0f;
                    pv[tt][i] = (val == 0.0f) ? NEG_INF : val;  // reference quirk
                }
            }

            // ---- online softmax ----
#pragma unroll
            for (int i = 0; i < 8; ++i) {
                float lm = fmaxf(pv[0][i], pv[1][i]);
#pragma unroll
                for (int off = 1; off < 16; off <<= 1)
                    lm = fmaxf(lm, __shfl_xor(lm, off, 32));
                float newm = fmaxf(mrun[i], lm);
                float scale = (mrun[i] == NEG_INF) ? 0.0f : __expf(mrun[i] - newm);
                float p0 = (pv[0][i] == NEG_INF) ? 0.0f : __expf(pv[0][i] - newm);
                float p1 = (pv[1][i] == NEG_INF) ? 0.0f : __expf(pv[1][i] - newm);
                pv[0][i] = p0;
                pv[1][i] = p1;
                float rs = p0 + p1;
#pragma unroll
                for (int off = 1; off < 16; off <<= 1)
                    rs += __shfl_xor(rs, off, 32);
                lrun[i] = lrun[i] * scale + rs;
                mrun[i] = newm;
#pragma unroll
                for (int t2 = 0; t2 < 4; ++t2) oacc[t2][i] *= scale;
            }

            // ---- restripe P (C-layout -> A-layout bf16) in per-wave LDS ----
#pragma unroll
            for (int tt = 0; tt < 2; ++tt)
#pragma unroll
                for (int i = 0; i < 8; ++i)
                    Ps[wave][hh * 8 + i][tt * 16 + n] = f32_to_bf16(pv[tt][i]);
            asm volatile("s_wait_dscnt 0" ::: "memory");

            FragBF pa;
            load_frag_a(pa, &Ps[wave][0][0], 40);
#pragma unroll
            for (int t2 = 0; t2 < 4; ++t2) {
                FragBF vb;   // B-frag of V == A-pattern read of V^T rows
                load_frag_a(vb, &VTs[buf][t2 * 16][0], 40);
                oacc[t2] = wmma_bf16(pa, vb, oacc[t2]);
            }
        }
        __syncthreads();   // all reads of buf done before its next TDM overwrite
    }

    // ---- normalize, write ctx [B,S,D] bf16 ----
#pragma unroll
    for (int i = 0; i < 8; ++i) {
        int qr = q0 + hh * 8 + i;
        float inv = 1.0f / lrun[i];
#pragma unroll
        for (int t2 = 0; t2 < 4; ++t2) {
            int d = t2 * 16 + n;
            ctx[((size_t)(b * S_ + qr)) * D_ + h * DK_ + d] =
                f32_to_bf16(oacc[t2][i] * inv);
        }
    }
}

// ---------------------------------------------------------------------------
// Kernel 3: output projection.  A tile (bf16 ctx) staged by TDM, double-
// buffered; Wo converted f32->bf16 on the VALU path into swizzled LDS.
// grid = (D/128, B*S/64), block = 256.
// ---------------------------------------------------------------------------
__global__ __launch_bounds__(256) void out_proj_kernel(
    const unsigned short* __restrict__ ctx, const float* __restrict__ Wo,
    const float* __restrict__ bo, float* __restrict__ out)
{
    __shared__ __align__(16) unsigned short As[2][64][40];
    __shared__ __align__(16) unsigned short Bsw[4096];

    const int m0 = blockIdx.y * 64;
    const int n0 = blockIdx.x * 128;
    const int tid = threadIdx.x;
    const int lane = tid & 31;
    const int wave = tid >> 5;
    const int wm = wave >> 2;
    const int wn = wave & 3;

    v8f acc[2][2];
#pragma unroll
    for (int i = 0; i < 2; ++i)
#pragma unroll
        for (int j = 0; j < 2; ++j) acc[i][j] = vzero();

    // A tile: 64 rows x 32 el; row 64B = 16 dw; pad 4 dw -> stride 40 ush
    if (wave == 0)
        tdm_load_2d(ctx + (size_t)m0 * D_, lds_offset(&As[0][0][0]),
                    D_, (unsigned)(B_ * S_), D_, 32, 64, 3u, 3u);

    for (int k0 = 0; k0 < D_; k0 += 32) {
        const int buf = (k0 >> 5) & 1;
        const bool has_next = (k0 + 32 < D_);
        if (wave == 0) {
            if (has_next) {
                tdm_load_2d(ctx + (size_t)m0 * D_ + k0 + 32, lds_offset(&As[buf ^ 1][0][0]),
                            D_, (unsigned)(B_ * S_), D_, 32, 64, 3u, 3u);
                wait_tensorcnt<1>();
            } else {
                wait_tensorcnt<0>();
            }
        }
        {   // Wo tile f32 -> bf16 swizzled (conversion path)
#pragma unroll
            for (int s2 = 0; s2 < 2; ++s2) {
                int seg = tid + s2 * 256;
                int n = seg & 127;
                int kseg = seg >> 7;
                const float* src = Wo + (size_t)(k0 + kseg * 8) * D_ + n0 + n;
                uint4 pk;
                pk.x = pack_bf16x2(src[0 * D_], src[1 * D_]);
                pk.y = pack_bf16x2(src[2 * D_], src[3 * D_]);
                pk.z = pack_bf16x2(src[4 * D_], src[5 * D_]);
                pk.w = pack_bf16x2(src[6 * D_], src[7 * D_]);
                *(uint4*)(Bsw + bsw_seg_base(n, kseg)) = pk;
                if (has_next) __builtin_prefetch(src + 32 * D_, 0, 3);
            }
        }
        __syncthreads();

        FragBF a0, a1, b0, b1;
        load_frag_a(a0, &As[buf][wm * 32][0], 40);
        load_frag_a(a1, &As[buf][wm * 32 + 16][0], 40);
        load_frag_b_sw(b0, Bsw + (wn * 2 + 0) * 512);
        load_frag_b_sw(b1, Bsw + (wn * 2 + 1) * 512);
        acc[0][0] = wmma_bf16(a0, b0, acc[0][0]);
        acc[0][1] = wmma_bf16(a0, b1, acc[0][1]);
        acc[1][0] = wmma_bf16(a1, b0, acc[1][0]);
        acc[1][1] = wmma_bf16(a1, b1, acc[1][1]);
        __syncthreads();
    }

    const int mbase = ((lane >> 4) << 3);
#pragma unroll
    for (int ni = 0; ni < 2; ++ni) {
        int nn = n0 + wn * 32 + ni * 16 + (lane & 15);
        float bias = bo[nn];
#pragma unroll
        for (int mi = 0; mi < 2; ++mi)
#pragma unroll
            for (int i = 0; i < 8; ++i) {
                int r = m0 + wm * 32 + mi * 16 + mbase + i;
                out[(size_t)r * D_ + nn] = acc[mi][ni][i] + bias;
            }
    }
}

// ---------------------------------------------------------------------------
// Launcher.  Workspace (bf16 elements): q | k | v^T | ctx, 4M each = 32MB.
// ---------------------------------------------------------------------------
extern "C" void kernel_launch(void* const* d_in, const int* in_sizes, int n_in,
                              void* d_out, int out_size, void* d_ws, size_t ws_size,
                              hipStream_t stream) {
    const float* Q    = (const float*)d_in[0];
    const float* K    = (const float*)d_in[1];
    const float* V    = (const float*)d_in[2];
    const int*   mask = (const int*)d_in[3];
    const float* Wq   = (const float*)d_in[4];
    const float* Wk   = (const float*)d_in[5];
    const float* Wv   = (const float*)d_in[6];
    const float* Wo   = (const float*)d_in[7];
    const float* bo   = (const float*)d_in[8];

    const size_t per = (size_t)B_ * H_ * S_ * DK_;  // 4M bf16 elements
    unsigned short* qws = (unsigned short*)d_ws;
    unsigned short* kws = qws + per;
    unsigned short* vws = kws + per;   // V^T
    unsigned short* cws = vws + per;

    dim3 g1(D_ / 128, (B_ * S_) / 64, 3);
    qkv_proj_kernel<<<g1, 256, 0, stream>>>(Q, K, V, Wq, Wk, Wv, qws, kws, vws);

    dim3 g2(S_ / 128, B_ * H_);
    attn_kernel<<<g2, 256, 0, stream>>>(qws, kws, vws, mask, cws);

    dim3 g3(D_ / 128, (B_ * S_) / 64);
    out_proj_kernel<<<g3, 256, 0, stream>>>(cws, Wo, bo, (float*)d_out);
}
